// GAUCLoss_25056839205782
// MI455X (gfx1250) — compile-verified, matrix operand order checked by reading, not run on hardware
//
#include <hip/hip_runtime.h>
#include <math.h>

#define NN 4096   // NUM_NODES
#define BB 2048   // BATCH
#define TT 2048   // T_TRAIN
#define WD 64     // WDIM
#define NC 10     // NUM_CLASSES

typedef _Float16 h16;
typedef __attribute__((ext_vector_type(16))) _Float16 v16h;
typedef __attribute__((ext_vector_type(8)))  _Float16 v8h;
typedef __attribute__((ext_vector_type(8)))  float    v8f;

// ---------------- workspace layout (bytes) ----------------
#define OFF_PSUB   ((size_t)0)                       // [BB][NN] fp16 : A*g_sub
#define OFF_PINT   (OFF_PSUB   + (size_t)BB*NN*2)    // [BB][NN] fp16 : A*g_inter
#define OFF_A01    (OFF_PINT   + (size_t)BB*NN*2)    // [BB][NN] fp16 : A (0/1)
#define OFF_GSUB   (OFF_A01    + (size_t)BB*NN*2)    // [NN] f32
#define OFF_GINT   (OFF_GSUB   + (size_t)NN*4)       // [NN] f32
#define OFF_WSS    (OFF_GINT   + (size_t)NN*4)       // [TT] f32
#define OFF_WSI    (OFF_WSS    + (size_t)TT*4)       // [TT] f32
#define OFF_ROWSUM (OFF_WSI    + (size_t)TT*4)       // [BB] f32
#define OFF_SDIAG  (OFF_ROWSUM + (size_t)BB*4)       // [BB] f32
#define OFF_INVN   (OFF_SDIAG  + (size_t)BB*4)       // [BB] f32
#define OFF_PART   (OFF_INVN   + (size_t)BB*4)       // [512] f32 block partials

// ---------------- LDS tile layout (bytes, padded stride, double buffered) ----------------
#define STRB     80                       // 64B row + 16B pad: 20 dwords, conflict-free
#define LDS_PSUB 0                        // 64 rows  x 32 halfs
#define LDS_PINT (64 * STRB)              // 64 rows
#define LDS_A01  (2 * 64 * STRB)          // 128 rows
#define LDS_BUF  (LDS_A01 + 128 * STRB)   // 20480 bytes per buffer

// 1) ws[t] = sum_d w_sub[d,t];  wi[t] = sum_d w_inter[d,t]
__global__ void k_wsum(const float* __restrict__ wsub, const float* __restrict__ wint,
                       float* __restrict__ outs, float* __restrict__ outi) {
    int t = blockIdx.x * blockDim.x + threadIdx.x;
    if (t >= TT) return;
    float a = 0.f, b = 0.f;
    for (int d = 0; d < WD; ++d) { a += wsub[d * TT + t]; b += wint[d * TT + t]; }
    outs[t] = a; outi[t] = b;
}

// 2) g_sub[k] = gem[k,:] . ws ; g_inter[k] = gem[k,:] . wi   (one block per k)
__global__ void k_g(const float* __restrict__ gem, const float* __restrict__ wss,
                    const float* __restrict__ wsi,
                    float* __restrict__ gsub, float* __restrict__ gint) {
    int k = blockIdx.x, tid = threadIdx.x;
    __shared__ float rs[256], ri[256];
    const float* row = gem + (size_t)k * TT;
    float a = 0.f, b = 0.f;
    for (int t = tid; t < TT; t += 256) { float g = row[t]; a += g * wss[t]; b += g * wsi[t]; }
    rs[tid] = a; ri[tid] = b; __syncthreads();
    for (int s = 128; s > 0; s >>= 1) {
        if (tid < s) { rs[tid] += rs[tid + s]; ri[tid] += ri[tid + s]; }
        __syncthreads();
    }
    if (tid == 0) { gsub[k] = rs[0]; gint[k] = ri[0]; }
}

// 3) pack fp16 GEMM operands
__global__ void k_pack(const unsigned char* __restrict__ adj, const int* __restrict__ mask_idx,
                       const float* __restrict__ gsub, const float* __restrict__ gint,
                       h16* __restrict__ psub, h16* __restrict__ pint, h16* __restrict__ a01) {
    size_t total = (size_t)BB * NN;
    for (size_t idx = (size_t)blockIdx.x * blockDim.x + threadIdx.x; idx < total;
         idx += (size_t)gridDim.x * blockDim.x) {
        int p = (int)(idx >> 12);
        int k = (int)(idx & (NN - 1));
        int row = mask_idx[p];
        float av = adj[(size_t)row * NN + k] ? 1.0f : 0.0f;
        a01[idx]  = (h16)av;
        psub[idx] = (h16)(av * gsub[k]);
        pint[idx] = (h16)(av * gint[k]);
    }
}

// 4) rowsum[p] = sum_k A[p,k]*g_sub[k]
__global__ void k_rowsum(const unsigned char* __restrict__ adj, const int* __restrict__ mask_idx,
                         const float* __restrict__ gsub, float* __restrict__ rowsum) {
    int p = blockIdx.x, tid = threadIdx.x;
    __shared__ float r[256];
    const unsigned char* ar = adj + (size_t)mask_idx[p] * NN;
    float a = 0.f;
    for (int k = tid; k < NN; k += 256) if (ar[k]) a += gsub[k];
    r[tid] = a; __syncthreads();
    for (int s = 128; s > 0; s >>= 1) { if (tid < s) r[tid] += r[tid + s]; __syncthreads(); }
    if (tid == 0) rowsum[p] = r[0];
}

// 5) class counts -> invN[b]; s[b] = pred[b, target[b]]
__global__ void k_batch(const float* __restrict__ pred, const int* __restrict__ target,
                        float* __restrict__ invN, float* __restrict__ sdiag) {
    __shared__ int cnt[NC];
    int tid = threadIdx.x;
    if (tid < NC) cnt[tid] = 0;
    __syncthreads();
    for (int b = tid; b < BB; b += blockDim.x) atomicAdd(&cnt[target[b]], 1);
    __syncthreads();
    for (int b = tid; b < BB; b += blockDim.x) {
        int t = target[b];
        invN[b]  = 1.0f / (float)cnt[t];
        sdiag[b] = pred[b * NC + t];
    }
}

// LDS fragment read per CDNA5 16-bit A/B layout:
// lane l: row = lrow0 + (l&15); chunk0 at halves c..c+7, chunk1 at c+16..c+23, c = 8*(l>=16)
__device__ __forceinline__ v16h lds_frag(const char* __restrict__ smem, int tileoff,
                                         int lrow0, int lane) {
    const char* p = smem + tileoff + (lrow0 + (lane & 15)) * STRB + ((lane >> 4) << 4);
    v8h lo = *(const v8h*)(p);
    v8h hi = *(const v8h*)(p + 32);
    return __builtin_shufflevector(lo, hi, 0,1,2,3,4,5,6,7,8,9,10,11,12,13,14,15);
}

#define WMMA(A, B, C) __builtin_amdgcn_wmma_f32_16x16x32_f16(false, (A), false, (B), (short)0, (C), false, false)

#define ASYNC_B128(ldsaddr, gptr)                                                   \
    asm volatile("global_load_async_to_lds_b128 %0, %1, off"                        \
                 :: "v"(ldsaddr), "v"((unsigned long long)(uintptr_t)(gptr)) : "memory")

// per-16x16-tile loss epilogue
__device__ __forceinline__ float epi_tile(const float* sS, const float* sI, int p0, int n0, int lane,
        const float* __restrict__ rowsum, const float* __restrict__ gsub,
        const unsigned char* __restrict__ adj, const int* __restrict__ mask_idx,
        const int* __restrict__ target, const float* __restrict__ pred,
        const float* __restrict__ sdiag, const float* __restrict__ invN) {
    int n     = n0 + (lane & 15);
    int mbase = (lane >> 4) << 3;
    int mn    = mask_idx[n];
    float gsn   = gsub[mn];
    float corrc = 1.0f - (adj[(size_t)mn * NN + mn] ? 1.0f : 0.0f);
    int   tn    = target[n];
    float invNn = invN[n];
    float lsum = 0.f;
#pragma unroll
    for (int r = 0; r < 8; ++r) {
        int p  = p0 + mbase + r;
        int tp = target[p];
        if (tp == tn) continue;
        float apn    = adj[(size_t)mask_idx[p] * NN + mn] ? 1.0f : 0.0f;
        float vi_sub = rowsum[p] - sS[r] - corrc * apn * gsn;
        float z  = (1.0f + vi_sub) / (1.0f + sI[r]);
        float v  = 1.0f - 1.0f / (1.0f + expf(-z));
        float dd = 1.0f - (sdiag[p] - pred[n * NC + tp]);
        lsum += invN[p] * invNn * v * dd * dd;
    }
    return lsum;
}

// 6) fused dual-GEMM, double-buffered async staging:
//    block = 256 thr = 8 waves (2m x 4n), block tile 64(m) x 128(n), wave tile 32x32.
//    Tile k+1 streams into LDS buffer (k+1)&1 via ASYNCcnt while WMMAs consume buffer k&1;
//    s_wait_asynccnt 0x4 tolerates the 4 in-flight next-tile loads (in-order completion).
__global__ void __launch_bounds__(256)
k_gemm(const h16* __restrict__ psub, const h16* __restrict__ pint, const h16* __restrict__ a01,
       const float* __restrict__ rowsum, const float* __restrict__ gsub,
       const unsigned char* __restrict__ adj, const int* __restrict__ mask_idx,
       const int* __restrict__ target, const float* __restrict__ pred,
       const float* __restrict__ sdiag, const float* __restrict__ invN,
       float* __restrict__ partial) {
    __shared__ char smem[2 * LDS_BUF];
    __shared__ float red[256];

    int tid  = threadIdx.x;
    int lane = tid & 31;
    int wave = tid >> 5;
    int p0b = blockIdx.y * 64;    // block m-range [p0b, p0b+64)
    int n0b = blockIdx.x * 128;   // block n-range [n0b, n0b+128)

    // ---- staging addressing: per thread one b128 chunk per 64-row tile ----
    int srow  = tid >> 2;              // 0..63
    int scol  = (tid & 3) * 8;         // half index {0,8,16,24}
    int sbyte = (tid & 3) * 16;        // byte offset in LDS row
    unsigned lds0  = (unsigned)(uintptr_t)smem;   // addrspace(3) offset in low bits
    unsigned ld_ps = lds0 + LDS_PSUB + srow * STRB + sbyte;
    unsigned ld_pi = lds0 + LDS_PINT + srow * STRB + sbyte;
    unsigned ld_b0 = lds0 + LDS_A01  + srow * STRB + sbyte;
    unsigned ld_b1 = ld_b0 + 64 * STRB;
    const h16* gb_ps = psub + (size_t)(p0b + srow) * NN + scol;       // k=0 base
    const h16* gb_pi = pint + (size_t)(p0b + srow) * NN + scol;
    const h16* gb_b0 = a01  + (size_t)(n0b + srow) * NN + scol;
    const h16* gb_b1 = a01  + (size_t)(n0b + 64 + srow) * NN + scol;

    // ---- wave tile position ----
    int wm = (wave >> 2) * 32;
    int wn = (wave & 3) * 32;

    v8f aS00 = {0.f,0.f,0.f,0.f,0.f,0.f,0.f,0.f}, aS01 = aS00, aS10 = aS00, aS11 = aS00;
    v8f aI00 = aS00, aI01 = aS00, aI10 = aS00, aI11 = aS00;

    // prologue: stream tile k=0 into buffer 0
    ASYNC_B128(ld_ps, gb_ps);
    ASYNC_B128(ld_pi, gb_pi);
    ASYNC_B128(ld_b0, gb_b0);
    ASYNC_B128(ld_b1, gb_b1);

#pragma unroll 1
    for (int kk = 0; kk < NN; kk += 32) {
        // issue next tile (wrapped on the last step; dead prefetch, drained at endpgm)
        int knext = (kk + 32) & (NN - 1);
        unsigned nb = (unsigned)(((kk >> 5) + 1) & 1) * LDS_BUF;
        ASYNC_B128(ld_ps + nb, gb_ps + knext);
        ASYNC_B128(ld_pi + nb, gb_pi + knext);
        ASYNC_B128(ld_b0 + nb, gb_b0 + knext);
        ASYNC_B128(ld_b1 + nb, gb_b1 + knext);

        // tolerate 4 in-flight (next tile); in-order completion => current tile resident
        asm volatile("s_wait_asynccnt 0x4" ::: "memory");
        __syncthreads();   // whole block's current tile visible

        const char* buf = smem + ((kk >> 5) & 1) * LDS_BUF;
        v16h as0 = lds_frag(buf, LDS_PSUB, wm,      lane);
        v16h as1 = lds_frag(buf, LDS_PSUB, wm + 16, lane);
        v16h ai0 = lds_frag(buf, LDS_PINT, wm,      lane);
        v16h ai1 = lds_frag(buf, LDS_PINT, wm + 16, lane);
        v16h b0  = lds_frag(buf, LDS_A01,  wn,      lane);
        v16h b1  = lds_frag(buf, LDS_A01,  wn + 16, lane);

        aS00 = WMMA(as0, b0, aS00);
        aS01 = WMMA(as0, b1, aS01);
        aS10 = WMMA(as1, b0, aS10);
        aS11 = WMMA(as1, b1, aS11);
        aI00 = WMMA(ai0, b0, aI00);
        aI01 = WMMA(ai0, b1, aI01);
        aI10 = WMMA(ai1, b0, aI10);
        aI11 = WMMA(ai1, b1, aI11);

        __syncthreads();   // all reads of this buffer done before it is re-filled
    }

    // ---- fused loss epilogue over the 2x2 tiles ----
    int p0 = p0b + wm, n0 = n0b + wn;
    float t0[8], t1[8], lsum = 0.f;

#pragma unroll
    for (int i = 0; i < 8; ++i) { t0[i] = aS00[i]; t1[i] = aI00[i]; }
    lsum += epi_tile(t0, t1, p0,      n0,      lane, rowsum, gsub, adj, mask_idx, target, pred, sdiag, invN);
#pragma unroll
    for (int i = 0; i < 8; ++i) { t0[i] = aS01[i]; t1[i] = aI01[i]; }
    lsum += epi_tile(t0, t1, p0,      n0 + 16, lane, rowsum, gsub, adj, mask_idx, target, pred, sdiag, invN);
#pragma unroll
    for (int i = 0; i < 8; ++i) { t0[i] = aS10[i]; t1[i] = aI10[i]; }
    lsum += epi_tile(t0, t1, p0 + 16, n0,      lane, rowsum, gsub, adj, mask_idx, target, pred, sdiag, invN);
#pragma unroll
    for (int i = 0; i < 8; ++i) { t0[i] = aS11[i]; t1[i] = aI11[i]; }
    lsum += epi_tile(t0, t1, p0 + 16, n0 + 16, lane, rowsum, gsub, adj, mask_idx, target, pred, sdiag, invN);

    red[tid] = lsum; __syncthreads();
    for (int s2 = 128; s2 > 0; s2 >>= 1) { if (tid < s2) red[tid] += red[tid + s2]; __syncthreads(); }
    if (tid == 0) partial[blockIdx.y * gridDim.x + blockIdx.x] = red[0];
}

// 7) deterministic final reduction
__global__ void k_reduce(const float* __restrict__ partial, int nb, float* __restrict__ out) {
    __shared__ float red[256];
    int tid = threadIdx.x;
    float a = 0.f;
    for (int i = tid; i < nb; i += 256) a += partial[i];
    red[tid] = a; __syncthreads();
    for (int s = 128; s > 0; s >>= 1) { if (tid < s) red[tid] += red[tid + s]; __syncthreads(); }
    if (tid == 0) out[0] = red[0];
}

extern "C" void kernel_launch(void* const* d_in, const int* in_sizes, int n_in,
                              void* d_out, int out_size, void* d_ws, size_t ws_size,
                              hipStream_t stream) {
    const float*         pred     = (const float*)d_in[0];
    const float*         gem      = (const float*)d_in[1];
    const float*         w_sub    = (const float*)d_in[2];
    const float*         w_inter  = (const float*)d_in[3];
    /* d_in[4] = w_global : unused */
    const unsigned char* adj      = (const unsigned char*)d_in[5];
    const int*           target   = (const int*)d_in[6];
    const int*           mask_idx = (const int*)d_in[7];

    char* ws = (char*)d_ws;
    h16*   psub   = (h16*)(ws + OFF_PSUB);
    h16*   pint   = (h16*)(ws + OFF_PINT);
    h16*   a01    = (h16*)(ws + OFF_A01);
    float* gsub   = (float*)(ws + OFF_GSUB);
    float* gint   = (float*)(ws + OFF_GINT);
    float* wss    = (float*)(ws + OFF_WSS);
    float* wsi    = (float*)(ws + OFF_WSI);
    float* rowsum = (float*)(ws + OFF_ROWSUM);
    float* sdiag  = (float*)(ws + OFF_SDIAG);
    float* invN   = (float*)(ws + OFF_INVN);
    float* part   = (float*)(ws + OFF_PART);

    k_wsum  <<<TT / 256, 256, 0, stream>>>(w_sub, w_inter, wss, wsi);
    k_g     <<<NN, 256, 0, stream>>>(gem, wss, wsi, gsub, gint);
    k_pack  <<<4096, 256, 0, stream>>>(adj, mask_idx, gsub, gint, psub, pint, a01);
    k_rowsum<<<BB, 256, 0, stream>>>(adj, mask_idx, gsub, rowsum);
    k_batch <<<1, 256, 0, stream>>>(pred, target, invN, sdiag);

    dim3 grid(BB / 128, BB / 64);   // 16 x 32 = 512 blocks
    k_gemm<<<grid, 256, 0, stream>>>(psub, pint, a01, rowsum, gsub, adj, mask_idx,
                                     target, pred, sdiag, invN, part);
    k_reduce<<<1, 256, 0, stream>>>(part, (int)(grid.x * grid.y), (float*)d_out);
}
